// CNN_3135326126769
// MI455X (gfx1250) — matrix operand order repaired and checked by reading, not compiled, not run on hardware
//
#include <hip/hip_runtime.h>
#include <hip/hip_bf16.h>
#include <cstdint>

typedef __attribute__((ext_vector_type(16))) _Float16 v16h;
typedef __attribute__((ext_vector_type(8)))  float    v8f;
typedef __attribute__((ext_vector_type(4)))  float    f4;

__device__ __forceinline__ v8f wmma16(v16h a, v16h b, v8f c) {
  return __builtin_amdgcn_wmma_f32_16x16x32_f16(false, a, false, b, (short)0, c, false, false);
}

// ---- CDNA5 async global->LDS copy (ASYNCcnt path) --------------------------
// The clang builtin exists but takes pointers in the un-spellable __device__ /
// __shared__ language address spaces, so issue the instruction directly.
// ISA: GLOBAL_LOAD_ASYNC_TO_LDS_B128  VDST(lds addr), VADDR(global addr), off
__device__ __forceinline__ void async_b128(const void* g, void* l) {
  uint32_t lo = (uint32_t)(uintptr_t)l;   // LDS byte offset = low 32 bits
  asm volatile("global_load_async_to_lds_b128 %0, %1, off"
               :: "v"(lo), "v"(g) : "memory");
}
__device__ __forceinline__ void async_wait0() {
#if __has_builtin(__builtin_amdgcn_s_wait_asynccnt)
  __builtin_amdgcn_s_wait_asynccnt(0);
#else
  asm volatile("s_wait_asynccnt 0x0" ::: "memory");
#endif
}

#define BATCH 32
#define HH 21
#define WW 180
#define HP 25
#define WP 184
#define SPAT (HH*WW)        /* 3780  */
#define PSPAT (HP*WP)       /* 4600  */
#define MTOT (BATCH*SPAT)   /* 120960 */
#define KFC 725760
#define NBLK_FC1 126
#define KCH (KFC/NBLK_FC1)  /* 5760 */

// ---------------------------------------------------------------------------
// small helpers
// ---------------------------------------------------------------------------
__global__ void zero_h16(_Float16* p, int n) {
  int i = blockIdx.x * blockDim.x + threadIdx.x;
  if (i < n) p[i] = (_Float16)0.f;
}

// pad fp32 input [32,1,21,180] -> f16 [32,1,25,184] with zero borders
__global__ void pad_input(const float* __restrict__ x, _Float16* __restrict__ xp) {
  int idx = blockIdx.x * blockDim.x + threadIdx.x;
  if (idx >= BATCH * PSPAT) return;
  int n = idx / PSPAT;
  int r = idx - n * PSPAT;
  int y = r / WP, xx = r - y * WP;
  float v = 0.f;
  if (y >= 2 && y < 2 + HH && xx >= 2 && xx < 2 + WW)
    v = x[(n * HH + (y - 2)) * WW + (xx - 2)];
  xp[idx] = (_Float16)v;
}

// pre-swizzle conv weights [COUT][CIN][5][5] fp32 into WMMA B-fragment layout:
// wfrag[kstep][co/16][lane][pos], lane=(co%16)+16*(kk/16), pos=kk%16, kk=k%32
template<int CIN, int COUT>
__global__ void prep_w(const float* __restrict__ w, _Float16* __restrict__ wfrag) {
  int idx = blockIdx.x * blockDim.x + threadIdx.x;
  constexpr int TOT = COUT * CIN * 25;
  if (idx >= TOT) return;
  int co = idx / (CIN * 25);
  int k  = idx - co * (CIN * 25);        // ci*25 + ky*5 + kx
  int ks = k >> 5, kk = k & 31;
  int nt = co >> 4;
  int lw = (co & 15) + ((kk & 16) ? 16 : 0);
  int pos = kk & 15;
  wfrag[(((size_t)ks * (COUT / 16) + nt) * 32 + lw) * 16 + pos] = (_Float16)w[idx];
}

// ---------------------------------------------------------------------------
// implicit-GEMM conv 5x5 SAME + bias + relu via WMMA f16->f32.
// Double-buffered LDS (one barrier per K-step); B tile staged with
// global_load_async_to_lds_b128 (ASYNCcnt), A tile im2col-gathered.
// ---------------------------------------------------------------------------
template<int CIN, int COUT, bool F32OUT>
__global__ __launch_bounds__(256)
void conv_wmma(const _Float16* __restrict__ in,
               const _Float16* __restrict__ wfrag,
               const float* __restrict__ bias,
               void* __restrict__ outv,
               int outCtot, int outCoff) {
  constexpr int K = CIN * 25;
  constexpr int KSTEPS = (K + 31) / 32;
  constexpr int NT = COUT / 16;
  __shared__ alignas(32) _Float16 As[2][8][32][16];
  __shared__ alignas(32) _Float16 Bs[2][NT][32][16];

  const int tid  = threadIdx.x;
  const int lane = tid & 31;
  const int wave = tid >> 5;
  const int mBase = blockIdx.x * 128;

  const v8f z8 = {0.f,0.f,0.f,0.f,0.f,0.f,0.f,0.f};
  v8f acc[NT];
#pragma unroll
  for (int t = 0; t < NT; ++t) acc[t] = z8;

  // A staging assignment: thread handles row arow, 16 consecutive k's
  const int arow = tid >> 1;
  const int m    = mBase + arow;
  const int n    = m / SPAT;
  const int rem  = m - n * SPAT;
  const int yy   = rem / WW;
  const int xx   = rem - yy * WW;
  const _Float16* inImg = in + (size_t)n * CIN * PSPAT;
  const int kHalf = (tid & 1) * 16;
  const int ar  = arow & 15;
  const int amt = arow >> 4;

  auto stageA = [&](int ksI, int buf) {
#pragma unroll
    for (int i = 0; i < 16; ++i) {
      const int kk = kHalf + i;
      const int kg = ksI * 32 + kk;
      _Float16 v = (_Float16)0.f;
      if (kg < K) {
        const int ci  = kg / 25;
        const int r25 = kg - ci * 25;
        const int ky  = r25 / 5;
        const int kx  = r25 - ky * 5;
        v = inImg[(ci * HP + yy + ky) * WP + (xx + kx)];
      }
      const int lw  = ar + ((kk & 16) ? 16 : 0);
      const int pos = kk & 15;
      As[buf][amt][lw][pos] = v;
    }
  };
  auto stageB = [&](int ksI, int buf) {
    const char* src = (const char*)(wfrag + (size_t)ksI * NT * 512);
    char* dstl = (char*)&Bs[buf][0][0][0];
#pragma unroll
    for (int i = 0; i < NT / 4; ++i) {
      const int c = (tid + i * 256) * 16;
      async_b128(src + c, dstl + c);
    }
  };

  // prologue: stage step 0 into buffer 0
  stageB(0, 0);
  stageA(0, 0);
  async_wait0();
  __syncthreads();

  for (int ks = 0; ks < KSTEPS; ++ks) {
    const int cur = ks & 1, nxt = cur ^ 1;
    if (ks + 1 < KSTEPS) stageB(ks + 1, nxt);     // async copy overlaps WMMA
    const v16h af = *(const v16h*)(&As[cur][wave][lane][0]);
#pragma unroll
    for (int t = 0; t < NT; ++t) {
      const v16h bf = *(const v16h*)(&Bs[cur][t][lane][0]);
      acc[t] = wmma16(af, bf, acc[t]);
    }
    if (ks + 1 < KSTEPS) stageA(ks + 1, nxt);
    async_wait0();
    __syncthreads();                              // one barrier per K-step
  }

  // ---- epilogue: bias + relu + scatter store ----
  const int col0 = lane & 15;
  const int rowH = (lane >> 4) * 8;
#pragma unroll
  for (int t = 0; t < NT; ++t) {
    const int co = t * 16 + col0;
    const float bv = bias[co];
#pragma unroll
    for (int r = 0; r < 8; ++r) {
      const int mm = mBase + wave * 16 + rowH + r;
      const int nn = mm / SPAT;
      const int rm = mm - nn * SPAT;
      const int y  = rm / WW;
      const int xq = rm - y * WW;
      float v = acc[t][r] + bv;
      v = v > 0.f ? v : 0.f;
      if constexpr (F32OUT) {
        float* outp = (float*)outv;
        outp[((size_t)(nn * outCtot + outCoff + co) * HH + y) * WW + xq] = v;
      } else {
        _Float16* outp = (_Float16*)outv;
        outp[((size_t)(nn * outCtot + outCoff + co) * HP + (y + 2)) * WP + (xq + 2)] = (_Float16)v;
      }
    }
  }
}

// concat helpers -------------------------------------------------------------
__global__ void copy_p3_to_p5(const _Float16* __restrict__ p3, _Float16* __restrict__ p5) {
  int idx = blockIdx.x * blockDim.x + threadIdx.x;
  const int per_n = 64 * PSPAT;
  if (idx >= BATCH * per_n) return;
  int n = idx / per_n;
  int r = idx - n * per_n;
  p5[((size_t)n * 128 + 64) * PSPAT + r] = p3[idx];
}

__global__ void copy_p2_to_x6(const _Float16* __restrict__ p2, float* __restrict__ x6) {
  int idx = blockIdx.x * blockDim.x + threadIdx.x;
  const int per_n = 64 * SPAT;
  if (idx >= BATCH * per_n) return;
  int n = idx / per_n;
  int r = idx - n * per_n;
  int c = r / SPAT;
  int rr = r - c * SPAT;
  int y = rr / WW, x = rr - y * WW;
  x6[((size_t)(n * 192 + 128 + c)) * SPAT + rr] =
      (float)p2[((size_t)(n * 64 + c) * HP + y + 2) * WP + (x + 2)];
}

// ---------------------------------------------------------------------------
// Hermite residual (lat) + spectral residual (lon, zero rfft k=0..2), in-place
// ---------------------------------------------------------------------------
__global__ __launch_bounds__(256)
void filter_kernel(float* __restrict__ x6) {
  __shared__ float plane[SPAT];
  __shared__ float phi[3][21];
  __shared__ float red[5][256];
  __shared__ float coef[5];
  const int tid = threadIdx.x;
  float* base = x6 + (size_t)blockIdx.x * SPAT;

  const float Lr = sqrtf(51.0f / 2.28e-11f);
  if (tid < 21) {
    const float lat = 20.f - 2.f * (float)tid;
    const float y = lat * 110000.f / Lr;
    const float e = expf(-y * y * 0.5f);
    const float p4 = sqrtf(sqrtf(3.14159265358979323846f)); // pi^(1/4)
    phi[0][tid] = e / p4;
    phi[1][tid] = e * 2.f * y / (1.41421356237f * p4);
    phi[2][tid] = e * (4.f * y * y - 2.f) / (2.82842712475f * p4);
  }
  for (int i = tid; i < SPAT; i += 256) plane[i] = base[i];
  __syncthreads();

  const float dy = 2.f * 110000.f / Lr;
  if (tid < WW) {
    float um0 = 0.f, um1 = 0.f, um2 = 0.f;
#pragma unroll
    for (int y = 0; y < 21; ++y) {
      float v = plane[y * WW + tid];
      um0 += v * phi[0][y]; um1 += v * phi[1][y]; um2 += v * phi[2][y];
    }
    um0 *= dy; um1 *= dy; um2 *= dy;
#pragma unroll
    for (int y = 0; y < 21; ++y)
      plane[y * WW + tid] -= um0 * phi[0][y] + um1 * phi[1][y] + um2 * phi[2][y];
  }
  __syncthreads();

  const float TWO_PI = 6.28318530717958647692f;
  const float th = TWO_PI * (float)tid / (float)WW;
  const float co1 = __cosf(th),       si1 = __sinf(th);
  const float co2 = __cosf(2.f * th), si2 = __sinf(2.f * th);
  for (int row = 0; row < 21; ++row) {
    float v = (tid < WW) ? plane[row * WW + tid] : 0.f;
    red[0][tid] = v;
    red[1][tid] = v * co1; red[2][tid] = v * si1;
    red[3][tid] = v * co2; red[4][tid] = v * si2;
    __syncthreads();
    for (int off = 128; off >= 1; off >>= 1) {
      if (tid < off) {
#pragma unroll
        for (int q = 0; q < 5; ++q) red[q][tid] += red[q][tid + off];
      }
      __syncthreads();
    }
    if (tid < 5) coef[tid] = red[tid][0];
    __syncthreads();
    if (tid < WW) {
      const float c0 = coef[0];
      const float c1r = coef[1], c1i = -coef[2];
      const float c2r = coef[3], c2i = -coef[4];
      const float rec = (c0 + 2.f * (c1r * co1 - c1i * si1)
                            + 2.f * (c2r * co2 - c2i * si2)) * (1.f / (float)WW);
      plane[row * WW + tid] -= rec;
    }
    __syncthreads();
  }
  for (int i = tid; i < SPAT; i += 256) base[i] = plane[i];
}

// ---------------------------------------------------------------------------
// FC1: [32 x 725760] @ [725760 x 500] -- memory bound on 1.45GB fp32 weights.
// Nontemporal weight streaming + in-register f32->f16 + WMMA; double-buffered
// LDS so next step's loads pipeline against current step's WMMAs.
// ---------------------------------------------------------------------------
__global__ __launch_bounds__(256)
void fc1_wmma(const float* __restrict__ act,     // [32][KFC] (= filtered x6)
              const float* __restrict__ w,       // [500][KFC]
              float* __restrict__ partial) {     // [NBLK_FC1][32][512]
  constexpr int STEPS = KCH / 32;                // 180
  __shared__ alignas(32) _Float16 As[2][2][32][16];
  __shared__ alignas(32) _Float16 Bs[2][32][32][16];
  const int tid = threadIdx.x, lane = tid & 31, wave = tid >> 5;
  const int kOff = blockIdx.x * KCH;

  const v8f z8 = {0.f,0.f,0.f,0.f,0.f,0.f,0.f,0.f};
  v8f acc[2][4];
#pragma unroll
  for (int mt = 0; mt < 2; ++mt)
#pragma unroll
    for (int t = 0; t < 4; ++t) acc[mt][t] = z8;

  const int ab  = tid >> 3;          // batch row 0..31
  const int akq = (tid & 7) * 4;     // k quad within 32
  const int alw = (ab & 15) + ((akq & 16) ? 16 : 0);
  const int amt = ab >> 4;
  const int apos = akq & 15;

  auto stage = [&](int s, int buf) {
    const int kb = kOff + s * 32;
    // ---- B: stream fp32 weights NT, convert, swizzle ----
#pragma unroll
    for (int jj = 0; jj < 2; ++jj) {
      const int j = tid * 2 + jj;
      v16h lo, hi;
      if (j < 500) {
        const f4* wr = (const f4*)(w + (size_t)j * KFC + kb);
        __builtin_prefetch(wr + 8, 0, 0);   // next K-step, same row
#pragma unroll
        for (int q = 0; q < 4; ++q) {
          const f4 f = __builtin_nontemporal_load(wr + q);
          lo[q * 4 + 0] = (_Float16)f.x; lo[q * 4 + 1] = (_Float16)f.y;
          lo[q * 4 + 2] = (_Float16)f.z; lo[q * 4 + 3] = (_Float16)f.w;
        }
#pragma unroll
        for (int q = 0; q < 4; ++q) {
          const f4 f = __builtin_nontemporal_load(wr + 4 + q);
          hi[q * 4 + 0] = (_Float16)f.x; hi[q * 4 + 1] = (_Float16)f.y;
          hi[q * 4 + 2] = (_Float16)f.z; hi[q * 4 + 3] = (_Float16)f.w;
        }
      } else {
#pragma unroll
        for (int q = 0; q < 16; ++q) { lo[q] = (_Float16)0.f; hi[q] = (_Float16)0.f; }
      }
      const int nt = j >> 4, jc = j & 15;
      *(v16h*)(&Bs[buf][nt][jc][0])      = lo;   // lanes 0..15 : K 0..15
      *(v16h*)(&Bs[buf][nt][jc + 16][0]) = hi;   // lanes 16..31: K 16..31
    }
    // ---- A: f32 activations -> f16 fragments ----
    {
      const f4 av = *(const f4*)(act + (size_t)ab * KFC + kb + akq);
      As[buf][amt][alw][apos + 0] = (_Float16)av.x;
      As[buf][amt][alw][apos + 1] = (_Float16)av.y;
      As[buf][amt][alw][apos + 2] = (_Float16)av.z;
      As[buf][amt][alw][apos + 3] = (_Float16)av.w;
    }
  };

  stage(0, 0);
  __syncthreads();

  for (int s = 0; s < STEPS; ++s) {
    const int cur = s & 1, nxt = cur ^ 1;
    const v16h a0 = *(const v16h*)(&As[cur][0][lane][0]);
    const v16h a1 = *(const v16h*)(&As[cur][1][lane][0]);
#pragma unroll
    for (int t = 0; t < 4; ++t) {
      const v16h bf = *(const v16h*)(&Bs[cur][wave * 4 + t][lane][0]);
      acc[0][t] = wmma16(a0, bf, acc[0][t]);
      acc[1][t] = wmma16(a1, bf, acc[1][t]);
    }
    if (s + 1 < STEPS) stage(s + 1, nxt);
    __syncthreads();                   // one barrier per K-step
  }

  // ---- write partials ----
  const int col0 = lane & 15, rowH = (lane >> 4) * 8;
  float* pb = partial + (size_t)blockIdx.x * 32 * 512;
#pragma unroll
  for (int mt = 0; mt < 2; ++mt)
#pragma unroll
    for (int t = 0; t < 4; ++t) {
      const int j = (wave * 4 + t) * 16 + col0;
#pragma unroll
      for (int r = 0; r < 8; ++r) {
        const int b = mt * 16 + rowH + r;
        pb[b * 512 + j] = acc[mt][t][r];
      }
    }
}

__global__ void fc1_reduce(const float* __restrict__ partial,
                           const float* __restrict__ bias,
                           float* __restrict__ h1) {
  int idx = blockIdx.x * blockDim.x + threadIdx.x;   // 32*512
  int b = idx >> 9, j = idx & 511;
  float s = 0.f;
  for (int p = 0; p < NBLK_FC1; ++p) s += partial[((size_t)p * 32 + b) * 512 + j];
  float v = (j < 500) ? s + bias[j] : 0.f;
  h1[idx] = v > 0.f ? v : 0.f;
}

// small dense layers ---------------------------------------------------------
__global__ void fc_dense(const float* __restrict__ in, const float* __restrict__ w,
                         const float* __restrict__ bias, float* __restrict__ out,
                         int B, int Kdim, int N, int inStride, int outStride, int doRelu) {
  int idx = blockIdx.x * blockDim.x + threadIdx.x;
  if (idx >= B * N) return;
  int b = idx / N, j = idx - b * N;
  const float* ip = in + (size_t)b * inStride;
  const float* wp = w + (size_t)j * Kdim;
  float s = bias[j];
  for (int k = 0; k < Kdim; ++k) s += ip[k] * wp[k];
  if (doRelu && s < 0.f) s = 0.f;
  out[(size_t)b * outStride + j] = s;
}

// ---------------------------------------------------------------------------
extern "C" void kernel_launch(void* const* d_in, const int* in_sizes, int n_in,
                              void* d_out, int out_size, void* d_ws, size_t ws_size,
                              hipStream_t stream) {
  (void)in_sizes; (void)n_in; (void)out_size; (void)ws_size;
  const float* x    = (const float*)d_in[0];
  const float* w_in = (const float*)d_in[1];
  const float* b_in = (const float*)d_in[2];
  const float* w1 = (const float*)d_in[3];  const float* b1 = (const float*)d_in[4];
  const float* w2 = (const float*)d_in[5];  const float* b2 = (const float*)d_in[6];
  const float* w3 = (const float*)d_in[7];  const float* b3 = (const float*)d_in[8];
  const float* w4 = (const float*)d_in[9];  const float* b4 = (const float*)d_in[10];
  const float* w5 = (const float*)d_in[11]; const float* b5 = (const float*)d_in[12];
  const float* wfc1 = (const float*)d_in[13]; const float* bfc1 = (const float*)d_in[14];
  const float* wfc2 = (const float*)d_in[15]; const float* bfc2 = (const float*)d_in[16];
  const float* wfc3 = (const float*)d_in[17]; const float* bfc3 = (const float*)d_in[18];
  const float* wfc4 = (const float*)d_in[19]; const float* bfc4 = (const float*)d_in[20];

  char* ws = (char*)d_ws;
  size_t off = 0;
  auto carve = [&](size_t bytes) -> char* {
    off = (off + 255) & ~(size_t)255;
    char* p = ws + off;
    off += bytes;
    return p;
  };
  _Float16* xpad = (_Float16*)carve((size_t)BATCH * PSPAT * 2);
  _Float16* wfin = (_Float16*)carve(32 * 64 * 2);
  _Float16* wfa1 = (_Float16*)carve(1600 * 64 * 2);
  _Float16* wfa2 = (_Float16*)carve(1600 * 64 * 2);
  _Float16* wfa3 = (_Float16*)carve(1600 * 64 * 2);
  _Float16* wfa4 = (_Float16*)carve(1600 * 64 * 2);
  _Float16* wfa5 = (_Float16*)carve(3200 * 128 * 2);
  _Float16* p1 = (_Float16*)carve((size_t)BATCH * 64 * PSPAT * 2);
  _Float16* p2 = (_Float16*)carve((size_t)BATCH * 64 * PSPAT * 2);
  _Float16* p3 = (_Float16*)carve((size_t)BATCH * 64 * PSPAT * 2);
  _Float16* p5 = (_Float16*)carve((size_t)BATCH * 128 * PSPAT * 2);
  float* x6      = (float*)carve((size_t)BATCH * 192 * SPAT * 4);
  float* partial = (float*)carve((size_t)NBLK_FC1 * 32 * 512 * 4);
  float* h1 = (float*)carve(32 * 512 * 4);
  float* h2 = (float*)carve(32 * 256 * 4);
  float* h3 = (float*)carve(32 * 64 * 4);
  _Float16* p4 = p1;   // x1 dead after conv1 -> reuse for x4

  // weight prep (pre-swizzled WMMA B fragments)
  zero_h16<<<8, 256, 0, stream>>>(wfin, 32 * 64);
  prep_w<1, 64><<<7, 256, 0, stream>>>(w_in, wfin);
  prep_w<64, 64><<<400, 256, 0, stream>>>(w1, wfa1);
  prep_w<64, 64><<<400, 256, 0, stream>>>(w2, wfa2);
  prep_w<64, 64><<<400, 256, 0, stream>>>(w3, wfa3);
  prep_w<64, 64><<<400, 256, 0, stream>>>(w4, wfa4);
  prep_w<128, 128><<<1600, 256, 0, stream>>>(w5, wfa5);
  pad_input<<<(BATCH * PSPAT + 255) / 256, 256, 0, stream>>>(x, xpad);

  // conv stack (implicit GEMM, M = 120960, 128 rows per block, 945 blocks)
  conv_wmma<1, 64, false><<<945, 256, 0, stream>>>(xpad, wfin, b_in, p1, 64, 0);
  conv_wmma<64, 64, false><<<945, 256, 0, stream>>>(p1, wfa1, b1, p2, 64, 0);
  conv_wmma<64, 64, false><<<945, 256, 0, stream>>>(p2, wfa2, b2, p3, 64, 0);
  conv_wmma<64, 64, false><<<945, 256, 0, stream>>>(p3, wfa3, b3, p4, 64, 0);
  conv_wmma<64, 64, false><<<945, 256, 0, stream>>>(p4, wfa4, b4, p5, 128, 0);
  copy_p3_to_p5<<<(BATCH * 64 * PSPAT + 255) / 256, 256, 0, stream>>>(p3, p5);
  conv_wmma<128, 128, true><<<945, 256, 0, stream>>>(p5, wfa5, b5, x6, 192, 0);
  copy_p2_to_x6<<<(BATCH * 64 * SPAT + 255) / 256, 256, 0, stream>>>(p2, x6);

  // Hermite + spectral residual, in place on x6
  filter_kernel<<<BATCH * 192, 256, 0, stream>>>(x6);

  // FC head
  fc1_wmma<<<NBLK_FC1, 256, 0, stream>>>(x6, wfc1, partial);
  fc1_reduce<<<64, 256, 0, stream>>>(partial, bfc1, h1);
  fc_dense<<<25, 256, 0, stream>>>(h1, wfc2, bfc2, h2, 32, 500, 200, 512, 256, 1);
  fc_dense<<<7, 256, 0, stream>>>(h2, wfc3, bfc3, h3, 32, 200, 50, 256, 64, 1);
  fc_dense<<<1, 256, 0, stream>>>(h3, wfc4, bfc4, (float*)d_out, 32, 50, 2, 64, 2, 0);
}
